// AttentionHead_83184926589394
// MI455X (gfx1250) — matrix operand order repaired and checked by reading, not compiled, not run on hardware
//
#include <hip/hip_runtime.h>
#include <hip/hip_bf16.h>
#include <stdint.h>

// ---------------------------------------------------------------------------
// Flash-attention (causal + reference pad-mask, 1/dk scaling) for MI455X
// (gfx1250): wave32, v_wmma_f32_16x16x32_bf16, LDS-staged K/V (64 keys/round),
// transposed-score formulation so softmax reductions stay register-local:
//   S^T[key][m] = K[key][:] . Q[m][:]            A = K tile,   B = Q^T
//   O^T[d][m]  += V^T[d][key] . P^T[key][m]      A = V^T tile, B = P^T
// ---------------------------------------------------------------------------

typedef __attribute__((ext_vector_type(16))) __bf16          v16bf;
typedef __attribute__((ext_vector_type(8)))  float           v8f;
typedef __attribute__((ext_vector_type(16))) unsigned short  u16x16;
typedef __attribute__((ext_vector_type(8)))  unsigned short  u16x8;
typedef __attribute__((ext_vector_type(4)))  float           f32x4;
typedef __attribute__((ext_vector_type(4)))  unsigned        u32x4;

#define S_LEN    4096
#define DK       64
#define BM       64          // query rows per workgroup (4 waves x 16 rows)
#define BKEYS    64          // keys staged to LDS per barrier round
#define NTHREADS 128
#define KSTRIDE  80          // LDS row stride (halfwords), 160B rows: 16B aligned
#define VSTRIDE  80

// scores enter softmax pre-scaled by log2(e)/dk so we can use raw v_exp_f32
#define SC_LOG2 (1.4426950408889634f / (float)DK)

#if __has_builtin(__builtin_amdgcn_exp2f)
#define EXP2F __builtin_amdgcn_exp2f
#else
#define EXP2F exp2f
#endif

// pack two floats to packed bf16 (round-to-nearest) in 3 VALU: 2x add + v_perm
__device__ __forceinline__ unsigned pack_bf16x2(float a, float b) {
  unsigned ua = __builtin_bit_cast(unsigned, a) + 0x8000u;
  unsigned ub = __builtin_bit_cast(unsigned, b) + 0x8000u;
#if __has_builtin(__builtin_amdgcn_perm)
  return __builtin_amdgcn_perm(ub, ua, 0x07060302u);  // {ub[3:2], ua[3:2]}
#else
  return (ua >> 16) | (ub & 0xFFFF0000u);
#endif
}

__device__ __forceinline__ v8f wmma_bf16(const u16x16& a, const u16x16& b, v8f c) {
  return __builtin_amdgcn_wmma_f32_16x16x32_bf16(
      /*neg_a=*/false, __builtin_bit_cast(v16bf, a),
      /*neg_b=*/false, __builtin_bit_cast(v16bf, b),
      /*c_mod=*/(short)0, c, /*reuse_a=*/false, /*reuse_b=*/false);
}

// A-fragment gather: two contiguous 16B runs at p and p+16B (K = base+0..15,
// contiguous order matching the B fragments)
__device__ __forceinline__ u16x16 ldsA(const unsigned short* p) {
  union { u16x8 h[2]; u16x16 v; } u;
  u.h[0] = *(const u16x8*)p;
  u.h[1] = *(const u16x8*)(p + 8);
  return u.v;
}

// one 32-key online-softmax step; MASK specialization chosen by a
// scalar-uniform branch so off-diagonal blocks pay zero causal-mask VALU.
template <bool MASK>
__device__ __forceinline__ void compute32(
    const unsigned short* __restrict__ sK,
    const unsigned short* __restrict__ sVT,
    int koff, int kg0, int qrow, int m, int h,
    const u16x16 qfrag[2], v8f acc[4], float& mi, float& li) {
  // ---- hoist all K-fragment LDS loads, then interleave the two WMMA chains
  const unsigned short* kr0 = &sK[(koff + m) * KSTRIDE + h * 16];
  const unsigned short* kr1 = &sK[(koff + 16 + m) * KSTRIDE + h * 16];
  u16x16 a00 = ldsA(kr0);        // tile 0, dk chunk 0
  u16x16 a01 = ldsA(kr0 + 32);   // tile 0, dk chunk 1
  u16x16 a10 = ldsA(kr1);        // tile 1, dk chunk 0
  u16x16 a11 = ldsA(kr1 + 32);   // tile 1, dk chunk 1
  v8f st[2];
  st[0] = wmma_bf16(a00, qfrag[0], (v8f){});
  st[1] = wmma_bf16(a10, qfrag[0], (v8f){});
  st[0] = wmma_bf16(a01, qfrag[1], st[0]);
  st[1] = wmma_bf16(a11, qfrag[1], st[1]);

  // ---- V^T fragments issued now: DS latency hides under softmax VALU ------
  u16x16 vf[4];
#pragma unroll
  for (int dt = 0; dt < 4; ++dt)
    vf[dt] = ldsA(&sVT[(dt * 16 + m) * VSTRIDE + koff + h * 16]);

  // ---- masks + online softmax (keys of query m live in lanes m, m+16) -----
  const float NEGINF = -__builtin_inff();
  float p[2][8];
  float bmax = NEGINF;
#pragma unroll
  for (int t = 0; t < 2; ++t)
#pragma unroll
    for (int r = 0; r < 8; ++r) {
      float raw = st[t][r];
      float sv = raw * SC_LOG2;              // scale by 1/dk (log2 domain)
      sv = (raw == 0.0f) ? NEGINF : sv;      // pad mask
      if (MASK) {
        const int key = kg0 + t * 16 + h * 8 + r;
        if (key > qrow) sv = NEGINF;         // causal mask
      }
      p[t][r] = sv;
      bmax = fmaxf(bmax, sv);
    }
  bmax = fmaxf(bmax, __shfl_xor(bmax, 16, 32));
  const float mn   = fmaxf(mi, bmax);
  const bool  dead = (mn == NEGINF);
  const float mnz  = dead ? 0.0f : mn;       // exp2(-inf - 0) == 0, no NaNs
  float alpha = EXP2F(mi - mnz);
  alpha = dead ? 1.0f : alpha;
  float rsum = 0.0f;
#pragma unroll
  for (int t = 0; t < 2; ++t)
#pragma unroll
    for (int r = 0; r < 8; ++r) {
      float pv = EXP2F(p[t][r] - mnz);
      p[t][r] = pv;
      rsum += pv;
    }
  rsum += __shfl_xor(rsum, 16, 32);
  li = li * alpha + rsum;
  mi = mn;
#pragma unroll
  for (int dt = 0; dt < 4; ++dt)
#pragma unroll
    for (int r = 0; r < 8; ++r) acc[dt][r] *= alpha;

  // ---- P^T B-fragment: pack bf16 + lane-half swap via shfl_xor(16) --------
  unsigned own[2][4], rcv[2][4];
#pragma unroll
  for (int t = 0; t < 2; ++t)
#pragma unroll
    for (int j = 0; j < 4; ++j)
      own[t][j] = pack_bf16x2(p[t][2 * j], p[t][2 * j + 1]);
#pragma unroll
  for (int t = 0; t < 2; ++t)
#pragma unroll
    for (int j = 0; j < 4; ++j)
      rcv[t][j] = (unsigned)__shfl_xor((int)own[t][j], 16, 32);
  union { unsigned d[8]; u16x16 v; } pf;
#pragma unroll
  for (int j = 0; j < 4; ++j) {
    pf.d[j]     = h ? rcv[1][j] : own[0][j];   // keys (h?16:0) .. +7
    pf.d[4 + j] = h ? own[1][j] : rcv[0][j];   // keys (h?24:8) .. +7
  }

  // ---- O^T += V^T . P^T : four 16(d)x16(m) accumulator tiles --------------
#pragma unroll
  for (int dt = 0; dt < 4; ++dt)
    acc[dt] = wmma_bf16(vf[dt], pf.v, acc[dt]);
}

__global__ __launch_bounds__(NTHREADS)
void AttentionHead_fa_wmma_kernel(const float* __restrict__ qg,
                                  const float* __restrict__ kg,
                                  const float* __restrict__ vg,
                                  float* __restrict__ outg) {
  __shared__ unsigned short sK[BKEYS * KSTRIDE];   // K tile, bf16, [key][d]
  __shared__ unsigned short sVT[DK * VSTRIDE];     // V^T tile, bf16, [d][key]

  const int tid  = threadIdx.x;
  const int wave = tid >> 5;
  const int lane = tid & 31;
  const int m    = lane & 15;
  const int h    = lane >> 4;

  const int blocks_per_b = S_LEN / BM;
  const int b      = blockIdx.x / blocks_per_b;
  const int qbase  = (blockIdx.x % blocks_per_b) * BM;
  const int qrow   = qbase + wave * 16 + m;
  // min query row of this wave, as a true scalar so the mask branch is SALU
  const int qw0    = __builtin_amdgcn_readfirstlane(qbase + wave * 16);

  const float* qB = qg + (size_t)b * S_LEN * DK;
  const float* kB = kg + (size_t)b * S_LEN * DK;
  const float* vB = vg + (size_t)b * S_LEN * DK;

  // ---- Q^T B-fragments (B 32x16: lane->col m, K = c*32 + h*16 + i) --------
  u16x16 qfrag[2];
#pragma unroll
  for (int c = 0; c < 2; ++c) {
    const f32x4* qp = (const f32x4*)(qB + (size_t)qrow * DK + c * 32 + h * 16);
    union { unsigned d[8]; u16x16 s; } f;
#pragma unroll
    for (int j = 0; j < 4; ++j) {
      f32x4 q4 = qp[j];
      f.d[2 * j]     = pack_bf16x2(q4[0], q4[1]);
      f.d[2 * j + 1] = pack_bf16x2(q4[2], q4[3]);
    }
    qfrag[c] = f.s;
  }

  v8f acc[4];
#pragma unroll
  for (int dt = 0; dt < 4; ++dt) acc[dt] = {};
  float mi = -__builtin_inff();
  float li = 0.0f;

  const int nkb = qbase / BKEYS + 1;   // 64-key tiles covering keys <= qbase+63

  // K staging: thread -> (key row, 32-float segment); fully coalesced
  const int skey = tid >> 1;
  const int sd0  = (tid & 1) * 32;
  // V^T staging: thread -> (d row, 32-key segment); per-instruction the 128
  // threads cover two full 256B V rows -> coalesced global, b128 LDS stores
  const int vd   = tid >> 1;
  const int vk0  = (tid & 1) * 32;

  for (int kb = 0; kb < nkb; ++kb) {
    __syncthreads();   // previous round's LDS reads complete

    // ---- stage K (row-major bf16) ----------------------------------------
    const float* kp = kB + (size_t)(kb * BKEYS + skey) * DK + sd0;
    {
      const f32x4* kpv = (const f32x4*)kp;
      unsigned kd[16];
#pragma unroll
      for (int j = 0; j < 8; ++j) {
        f32x4 k4 = kpv[j];
        kd[2 * j]     = pack_bf16x2(k4[0], k4[1]);
        kd[2 * j + 1] = pack_bf16x2(k4[2], k4[3]);
      }
      u32x4* dst = (u32x4*)&sK[skey * KSTRIDE + sd0];
#pragma unroll
      for (int j = 0; j < 4; ++j) {
        u32x4 t4;
#pragma unroll
        for (int q4 = 0; q4 < 4; ++q4) t4[q4] = kd[4 * j + q4];
        dst[j] = t4;
      }
    }
    // ---- stage V^T (transposed bf16) -------------------------------------
    {
      const float* vp = vB + (size_t)(kb * BKEYS + vk0) * DK + vd;
      unsigned vdw[16];
#pragma unroll
      for (int j = 0; j < 16; ++j) {
        float a = vp[(2 * j) * DK];
        float c = vp[(2 * j + 1) * DK];
        vdw[j] = pack_bf16x2(a, c);
      }
      u32x4* dst = (u32x4*)&sVT[vd * VSTRIDE + vk0];
#pragma unroll
      for (int j = 0; j < 4; ++j) {
        u32x4 t4;
#pragma unroll
        for (int q4 = 0; q4 < 4; ++q4) t4[q4] = vdw[4 * j + q4];
        dst[j] = t4;
      }
    }
    if (kb + 1 < nkb) {   // global_prefetch_b8 next K and V tiles
      __builtin_prefetch(kp + BKEYS * DK, 0, 1);
      __builtin_prefetch(vB + (size_t)((kb + 1) * BKEYS + skey) * DK + sd0, 0, 1);
    }
    __syncthreads();

    // ---- two 32-key compute steps; causal mask only on diagonal blocks ----
#pragma unroll
    for (int half = 0; half < 2; ++half) {
      const int koff = half * 32;
      const int kg0  = kb * BKEYS + koff;
      if (kg0 + 31 > qw0)   // scalar branch
        compute32<true >(sK, sVT, koff, kg0, qrow, m, h, qfrag, acc, mi, li);
      else
        compute32<false>(sK, sVT, koff, kg0, qrow, m, h, qfrag, acc, mi, li);
    }
  }

  // ---- Epilogue: O[m][d] = O^T[d][m] / li (0 for fully-masked rows) -------
  const float inv = (li > 0.0f && li == li) ? (1.0f / li) : 0.0f;
  float* op = outg + ((size_t)b * S_LEN + qrow) * DK;
#pragma unroll
  for (int dt = 0; dt < 4; ++dt) {
    f32x4 o0, o1;
#pragma unroll
    for (int r = 0; r < 4; ++r) {
      o0[r] = acc[dt][r] * inv;
      o1[r] = acc[dt][4 + r] * inv;
    }
    float* dst = op + dt * 16 + h * 8;
    *(f32x4*)dst       = o0;
    *(f32x4*)(dst + 4) = o1;
  }
}

extern "C" void kernel_launch(void* const* d_in, const int* in_sizes, int n_in,
                              void* d_out, int out_size, void* d_ws, size_t ws_size,
                              hipStream_t stream) {
  const float* q = (const float*)d_in[0];
  const float* k = (const float*)d_in[1];
  const float* v = (const float*)d_in[2];
  float* out = (float*)d_out;

  const int B = in_sizes[0] / (S_LEN * DK);           // = 8
  const int nblocks = B * (S_LEN / BM);               // = 512 workgroups
  AttentionHead_fa_wmma_kernel<<<nblocks, NTHREADS, 0, stream>>>(q, k, v, out);
}
